// MultiHeadAttention_68702296867207
// MI455X (gfx1250) — matrix operand order repaired and checked by reading, not compiled, not run on hardware
//
#include <hip/hip_runtime.h>

// Problem constants (match reference)
#define BB 2
#define SS 2048
#define DD 1024
#define HH 16
#define DH 64
#define LDQ (2 * DD)   // qkvh row stride: [q | k] only; v goes to vT buffer

typedef __attribute__((ext_vector_type(16))) _Float16 v16h;
typedef __attribute__((ext_vector_type(8)))  float    v8f;

struct alignas(16) H8 { _Float16 h[8]; };

// ---------------------------------------------------------------------------
// CDNA5 async copy: global -> LDS, 16B per lane, tracked by ASYNCcnt.
// ---------------------------------------------------------------------------
__device__ __forceinline__ void async_copy_b128(const _Float16* gsrc,
                                                const _Float16* ldst) {
  unsigned lds = (unsigned)(size_t)ldst;               // low 32 bits = LDS addr
  unsigned long long ga = (unsigned long long)(size_t)gsrc;
  asm volatile("global_load_async_to_lds_b128 %0, %1, off"
               :: "v"(lds), "v"(ga) : "memory");
}
__device__ __forceinline__ void wait_async0() {
  asm volatile("s_wait_asynccnt 0" ::: "memory");
}

// ---------------------------------------------------------------------------
// WMMA wrapper: D = A(16x32 f16) * B(32x16 f16) + C(16x16 f32)
// ---------------------------------------------------------------------------
__device__ __forceinline__ v8f wmma16(v16h a, v16h b, v8f c) {
  return __builtin_amdgcn_wmma_f32_16x16x32_f16(
      false, a, false, b, (short)0, c, false, false);
}

// A-fragment (16x32 MxK) from row-major [row][k], stride ld.
// lane = half*16 + m; elem e -> K = (e/8)*16 + half*8 + (e%8)
__device__ __forceinline__ v16h load_a_frag(const _Float16* base, int ld) {
  const int lane = threadIdx.x & 31;
  const int hlf = lane >> 4, m = lane & 15;
  const _Float16* p = base + m * ld;
  v16h a;
#pragma unroll
  for (int i = 0; i < 8; ++i) a[i] = p[hlf * 8 + i];
#pragma unroll
  for (int i = 0; i < 8; ++i) a[8 + i] = p[16 + hlf * 8 + i];
  return a;
}

// B-fragment (32x16 KxN) from TRANSPOSED storage bt[n][k], stride ld.
// lane = half*16 + n; elem e -> K = (e/8)*16 + half*8 + (e%8)
__device__ __forceinline__ v16h load_bt_frag(const _Float16* bt, int ld) {
  const int lane = threadIdx.x & 31;
  const int hlf = lane >> 4, n = lane & 15;
  const _Float16* p = bt + n * ld + hlf * 8;
  v16h b;
#pragma unroll
  for (int e = 0; e < 16; ++e) b[e] = p[e < 8 ? e : e + 8];
  return b;
}

// ---------------------------------------------------------------------------
// fp32 -> f16 conversion (x4)
// ---------------------------------------------------------------------------
__global__ void cvt_f32_to_f16(const float* __restrict__ in,
                               _Float16* __restrict__ out, int n) {
  int i = (blockIdx.x * blockDim.x + threadIdx.x) * 4;
  if (i < n) {
    float4 v = *reinterpret_cast<const float4*>(in + i);
    out[i + 0] = (_Float16)v.x;
    out[i + 1] = (_Float16)v.y;
    out[i + 2] = (_Float16)v.z;
    out[i + 3] = (_Float16)v.w;
  }
}

// fp32 [K][N] -> f16 transposed [N][K]; 32x32 tiles via LDS, coalesced both ways
__global__ void cvt_transpose_f16(const float* __restrict__ in,
                                  _Float16* __restrict__ out, int K, int N) {
  __shared__ float tile[32][33];
  const int k0 = blockIdx.y * 32, n0 = blockIdx.x * 32;
  const int tx = threadIdx.x & 31, ty = threadIdx.x >> 5;  // 8 row-groups
#pragma unroll
  for (int i = 0; i < 4; ++i)
    tile[ty + i * 8][tx] = in[(size_t)(k0 + ty + i * 8) * N + n0 + tx];
  __syncthreads();
#pragma unroll
  for (int i = 0; i < 4; ++i)
    out[(size_t)(n0 + ty + i * 8) * K + k0 + tx] = (_Float16)tile[tx][ty + i * 8];
}

// ---------------------------------------------------------------------------
// Tiled WMMA GEMM with async-to-LDS double buffering.
// C[M,N] = A[M,K](f16) * B[K,N] + bias[N], B given TRANSPOSED as Bt[N][K].
// WG = 256 threads (8 waves); tile 64x128; wave owns 32x32 (4 accumulators).
// If vt != nullptr, columns gn >= 2*DD are routed to vT[b,h,dh,s] instead
// (used by the QKV projection to emit V pre-transposed for attention).
// ---------------------------------------------------------------------------
#define TM 64
#define TN 128
#define TK 32
#define SA_LD (TK + 8)   // 40 f16 rows -> 80B (16B aligned), bank-spreading
#define SB_LD (TK + 8)

template <typename OutT>
__global__ void gemm_wmma(const _Float16* __restrict__ A,
                          const _Float16* __restrict__ Bt,
                          const float* __restrict__ bias,
                          OutT* __restrict__ C, _Float16* __restrict__ vt,
                          int M, int N, int K, int ldc) {
  __shared__ alignas(16) _Float16 sA[2][TM][SA_LD];
  __shared__ alignas(16) _Float16 sBt[2][TN][SB_LD];

  const int wave = threadIdx.x >> 5;
  const int lane = threadIdx.x & 31;
  const int wm = wave >> 2, wn = wave & 3;
  const int m0 = blockIdx.y * TM;
  const int n0 = blockIdx.x * TN;

  auto issue = [&](int buf, int k0) {
    const int t = threadIdx.x;
    {  // A tile: 64 rows x 32 f16, 4 threads/row x 16B
      const int r = t >> 2, c = (t & 3) * 8;
      async_copy_b128(A + (size_t)(m0 + r) * K + k0 + c, &sA[buf][r][c]);
    }
    {  // Bt tile: 128 rows x 32 f16, 2 threads/row x 2x16B
      const int n = t >> 1, c = (t & 1) * 16;
      const _Float16* g = Bt + (size_t)(n0 + n) * K + k0 + c;
      async_copy_b128(g, &sBt[buf][n][c]);
      async_copy_b128(g + 8, &sBt[buf][n][c + 8]);
    }
  };

  v8f acc[2][2] = {};
  issue(0, 0);
  for (int k0 = 0; k0 < K; k0 += TK) {
    const int cur = (k0 / TK) & 1;
    wait_async0();
    __syncthreads();
    if (k0 + TK < K) issue(cur ^ 1, k0 + TK);   // prefetch next tile

#pragma unroll
    for (int i = 0; i < 2; ++i) {
      v16h af = load_a_frag(&sA[cur][wm * 32 + i * 16][0], SA_LD);
#pragma unroll
      for (int j = 0; j < 2; ++j) {
        v16h bf = load_bt_frag(&sBt[cur][wn * 32 + j * 16][0], SB_LD);
        acc[i][j] = wmma16(af, bf, acc[i][j]);
      }
    }
    __syncthreads();
  }

  const int hlf = lane >> 4, n = lane & 15;
#pragma unroll
  for (int i = 0; i < 2; ++i) {
#pragma unroll
    for (int j = 0; j < 2; ++j) {
      const int gm = m0 + wm * 32 + i * 16 + hlf * 8;
      const int gn = n0 + wn * 32 + j * 16 + n;
      const float bv = bias[gn];
      if (vt && gn >= 2 * DD) {
        // V columns -> vT[b][h][dh][s], 8 consecutive s per lane = one 16B store
        const int d = gn - 2 * DD;
        const int h = d >> 6, dh = d & 63;
        const int b = gm >> 11, s0 = gm & (SS - 1);
        H8 o;
#pragma unroll
        for (int r = 0; r < 8; ++r) o.h[r] = (_Float16)(acc[i][j][r] + bv);
        *reinterpret_cast<H8*>(vt + (((size_t)(b * HH + h) * DH + h * 0 + dh) * SS + s0)) = o;
      } else {
#pragma unroll
        for (int r = 0; r < 8; ++r)
          C[(size_t)(gm + r) * ldc + gn] = (OutT)(acc[i][j][r] + bv);
      }
    }
  }
}

// ---------------------------------------------------------------------------
// Flash attention, causal + forcing folded into exp weights (log2 domain).
// Grid: x = S/128, y = B*H. 256 threads = 8 waves x 16 q rows.
// K/V tiles arrive via async-to-LDS with double buffering.
// ---------------------------------------------------------------------------
__global__ void attn_flash(const _Float16* __restrict__ qkvh,  // [B*S][2D]: q|k
                           const _Float16* __restrict__ vt,    // [B*H*DH][S]
                           const int* __restrict__ idxp,
                           const float* __restrict__ wptr,
                           _Float16* __restrict__ ctx) {       // [B*S][D]
  __shared__ alignas(16) _Float16 sK[2][32][DH + 8];   // [kv][dh] == Bt for QK^T
  __shared__ alignas(16) _Float16 sVt[2][DH][32 + 8];  // [dh][kv] == Bt for PV
  __shared__ alignas(16) _Float16 sP[8][16][32 + 8];   // per-wave P bounce

  const int bh = blockIdx.y;
  const int b = bh >> 4, h = bh & (HH - 1);
  const int q0wg = blockIdx.x * 128;
  const int wave = threadIdx.x >> 5;
  const int lane = threadIdx.x & 31;
  const int hlf = lane >> 4, nn = lane & 15;
  const int qw = q0wg + wave * 16;

  const float wforce = *wptr;
  const int fidx = idxp[b];

  const _Float16* qbase = qkvh + (size_t)b * SS * LDQ + h * DH;
  const _Float16* kbase = qbase + DD;
  const _Float16* vtbase = vt + (size_t)(b * HH + h) * DH * SS;

  // Q fragments; fold 1/sqrt(dh) * log2(e) so scores are in log2 domain
  const float qscale = 0.125f * 1.44269504088896f;
  v16h qf[2];
  {
    const _Float16* p = qbase + (size_t)(qw + nn) * LDQ;
#pragma unroll
    for (int t = 0; t < 2; ++t) {
#pragma unroll
      for (int i = 0; i < 8; ++i)
        qf[t][i] = (_Float16)((float)p[t * 32 + hlf * 8 + i] * qscale);
#pragma unroll
      for (int i = 0; i < 8; ++i)
        qf[t][8 + i] = (_Float16)((float)p[t * 32 + 16 + hlf * 8 + i] * qscale);
    }
  }

  auto issue = [&](int buf, int kv0) {
    const int t = threadIdx.x;
    {  // K tile: 32 rows x 64 f16
      const int r = t >> 3, c = (t & 7) * 8;
      async_copy_b128(kbase + (size_t)(kv0 + r) * LDQ + c, &sK[buf][r][c]);
    }
    {  // Vt tile: 64 rows x 32 f16 (straight copy thanks to vT layout)
      const int r = t >> 2, c = (t & 3) * 8;
      async_copy_b128(vtbase + (size_t)r * SS + kv0 + c, &sVt[buf][r][c]);
    }
  };

  float mrow[8], lrow[8];
#pragma unroll
  for (int r = 0; r < 8; ++r) { mrow[r] = -1e30f; lrow[r] = 0.0f; }
  v8f oacc[4] = {};

  const int kv_end = q0wg + 128;
  issue(0, 0);
  for (int kv0 = 0; kv0 < kv_end; kv0 += 32) {
    const int cur = (kv0 >> 5) & 1;
    wait_async0();
    __syncthreads();
    if (kv0 + 32 < kv_end) issue(cur ^ 1, kv0 + 32);

    if (kv0 < qw + 16) {  // wave-uniform; EXEC stays all-ones for WMMA
      v8f s[2] = {};
#pragma unroll
      for (int ct = 0; ct < 2; ++ct)
#pragma unroll
        for (int t = 0; t < 2; ++t) {
          v16h bf = load_bt_frag(&sK[cur][ct * 16][t * 32], DH + 8);
          s[ct] = wmma16(qf[t], bf, s[ct]);
        }

#pragma unroll
      for (int ct = 0; ct < 2; ++ct)
#pragma unroll
        for (int r = 0; r < 8; ++r) {
          const int qi = qw + hlf * 8 + r;
          const int kj = kv0 + ct * 16 + nn;
          if (kj > qi) s[ct][r] = -1e30f;
        }

      v8f mt;
#pragma unroll
      for (int r = 0; r < 8; ++r) mt[r] = fmaxf(s[0][r], s[1][r]);
#pragma unroll
      for (int off = 1; off < 16; off <<= 1)
#pragma unroll
        for (int r = 0; r < 8; ++r)
          mt[r] = fmaxf(mt[r], __shfl_xor(mt[r], off, 32));

      float osc[8];
#pragma unroll
      for (int r = 0; r < 8; ++r) {
        const float mn = fmaxf(mrow[r], mt[r]);
        osc[r] = __builtin_amdgcn_exp2f(mrow[r] - mn);
        mrow[r] = mn;
      }

      const float a0 = (kv0 + nn >= fidx) ? wforce : 1.0f;
      const float a1 = (kv0 + 16 + nn >= fidx) ? wforce : 1.0f;
      v8f p0, p1, rs;
#pragma unroll
      for (int r = 0; r < 8; ++r) {
        p0[r] = __builtin_amdgcn_exp2f(s[0][r] - mrow[r]) * a0;
        p1[r] = __builtin_amdgcn_exp2f(s[1][r] - mrow[r]) * a1;
        rs[r] = p0[r] + p1[r];
      }
#pragma unroll
      for (int off = 1; off < 16; off <<= 1)
#pragma unroll
        for (int r = 0; r < 8; ++r) rs[r] += __shfl_xor(rs[r], off, 32);
#pragma unroll
      for (int r = 0; r < 8; ++r) lrow[r] = lrow[r] * osc[r] + rs[r];

#pragma unroll
      for (int c4 = 0; c4 < 4; ++c4)
#pragma unroll
        for (int r = 0; r < 8; ++r) oacc[c4][r] *= osc[r];

#pragma unroll
      for (int r = 0; r < 8; ++r) {
        sP[wave][hlf * 8 + r][nn]      = (_Float16)p0[r];
        sP[wave][hlf * 8 + r][16 + nn] = (_Float16)p1[r];
      }
      v16h pf = load_a_frag(&sP[wave][0][0], 32 + 8);
#pragma unroll
      for (int c4 = 0; c4 < 4; ++c4) {
        v16h vf = load_bt_frag(&sVt[cur][c4 * 16][0], 32 + 8);
        oacc[c4] = wmma16(pf, vf, oacc[c4]);
      }
    }
    __syncthreads();
  }

#pragma unroll
  for (int r = 0; r < 8; ++r) {
    const float inv = 1.0f / lrow[r];
    const size_t row = (size_t)b * SS + qw + hlf * 8 + r;
#pragma unroll
    for (int c4 = 0; c4 < 4; ++c4)
      ctx[row * DD + h * DH + c4 * 16 + nn] = (_Float16)(oacc[c4][r] * inv);
  }
}

// ---------------------------------------------------------------------------
// Launch
// ---------------------------------------------------------------------------
extern "C" void kernel_launch(void* const* d_in, const int* in_sizes, int n_in,
                              void* d_out, int out_size, void* d_ws, size_t ws_size,
                              hipStream_t stream) {
  const float* x      = (const float*)d_in[0];
  const int*   idx    = (const int*)d_in[1];
  const float* weight = (const float*)d_in[2];
  // d_in[3] = attn_forcing (static 1 in reference)
  const float* w_qkv  = (const float*)d_in[4];
  const float* b_qkv  = (const float*)d_in[5];
  const float* w_proj = (const float*)d_in[6];
  const float* b_proj = (const float*)d_in[7];
  float* out = (float*)d_out;

  const size_t nX    = (size_t)BB * SS * DD;      // 4194304
  const size_t nWqkv = (size_t)DD * 3 * DD;       // 3145728
  const size_t nWprj = (size_t)DD * DD;           // 1048576
  const size_t nQK   = (size_t)BB * SS * 2 * DD;  // 8388608 (q|k only)
  const size_t nVT   = (size_t)BB * HH * DH * SS; // 4194304

  _Float16* xh   = (_Float16*)d_ws;
  _Float16* wqt  = xh + nX;          // w_qkv^T [3D][D]
  _Float16* wpt  = wqt + nWqkv;      // w_proj^T [D][D]
  _Float16* qkvh = wpt + nWprj;      // [B*S][2D] q|k
  _Float16* vth  = qkvh + nQK;       // V^T [B*H*DH][S]
  _Float16* ctxh = vth + nVT;        // [B*S][D]

  cvt_f32_to_f16<<<(int)(nX / 1024), 256, 0, stream>>>(x, xh, (int)nX);
  cvt_transpose_f16<<<dim3(3 * DD / 32, DD / 32), 256, 0, stream>>>(w_qkv, wqt, DD, 3 * DD);
  cvt_transpose_f16<<<dim3(DD / 32, DD / 32), 256, 0, stream>>>(w_proj, wpt, DD, DD);

  // QKV projection; V third routed to vT
  gemm_wmma<_Float16><<<dim3(3 * DD / TN, BB * SS / TM), 256, 0, stream>>>(
      xh, wqt, b_qkv, qkvh, vth, BB * SS, 3 * DD, DD, LDQ);

  attn_flash<<<dim3(SS / 128, BB * HH), 256, 0, stream>>>(qkvh, vth, idx, weight, ctxh);

  gemm_wmma<float><<<dim3(DD / TN, BB * SS / TM), 256, 0, stream>>>(
      ctxh, wpt, b_proj, out, nullptr, BB * SS, DD, DD, DD);
}
